// StructuredReasoner_75316546503113
// MI455X (gfx1250) — matrix operand order repaired and checked by reading, not compiled, not run on hardware
//
#include <hip/hip_runtime.h>

// ---------------------------------------------------------------------------
// StructuredReasoner for MI455X (gfx1250): bf16 WMMA (f32 accumulate) GEMMs,
// memory-bound by design (~280 MB HBM traffic, ~12 us floor @ 23.3 TB/s).
// ---------------------------------------------------------------------------

typedef __attribute__((ext_vector_type(16))) __bf16 v16bf;
typedef __attribute__((ext_vector_type(8)))  float  v8f;
typedef unsigned short u16;

#define BTOK 16384   // B*T
#define DIM  2048
#define RR   128
#define EE   8
#define HH   512

// workspace layout (bytes)
#define WS_VPACK   ((size_t)0)
#define WS_UPACK   (WS_VPACK  + 512u*1024u)      // Vpack 512 KB
#define WS_W1PACK  (WS_UPACK  + 512u*1024u)      // Upack 512 KB
#define WS_W2PACK  (WS_W1PACK + 1024u*1024u)     // W1pack 1 MB
#define WS_WBUF    (WS_W2PACK + 1024u*1024u)     // W2pack 1 MB
#define WS_BLEND   (WS_WBUF   + (size_t)BTOK*EE*4u)

// output layout (floats): h_new, probs, p_halt, z_final
#define OUT_HNEW  ((size_t)0)
#define OUT_PROBS ((size_t)BTOK*DIM)
#define OUT_PHALT (OUT_PROBS + (size_t)BTOK*EE)
#define OUT_ZF    (OUT_PHALT + (size_t)BTOK)

struct __attribute__((aligned(16))) U4 { unsigned int x, y, z, w; };
struct __attribute__((aligned(16))) F4 { float x, y, z, w; };

__device__ __forceinline__ u16 f2bf(float f) {  // round-to-nearest-even
  unsigned u = __float_as_uint(f);
  u += 0x7FFFu + ((u >> 16) & 1u);
  return (u16)(u >> 16);
}

union Frag { v16bf v; U4 q[2]; u16 s[16]; };

__device__ __forceinline__ v8f wmma_bf16(const Frag& a, const Frag& b, v8f c) {
  return __builtin_amdgcn_wmma_f32_16x16x32_bf16(false, a.v, false, b.v,
                                                 (short)0, c, false, false);
}

__device__ __forceinline__ v8f v8f_zero() {
  v8f z;
#pragma unroll
  for (int j = 0; j < 8; ++j) z[j] = 0.f;
  return z;
}

// ---------------------------------------------------------------------------
// Pack a [K x N] f32 matrix (batched) into WMMA bf16 B-fragment layout:
// frag (b, nt, kt): lane<16  -> col N = nt*16+lane,    K = kt*32 + 0..7, 16..23
//                   lane>=16 -> col N = nt*16+lane-16, K = kt*32 + 8..15, 24..31
// 32 bytes per lane, contiguous per fragment -> 2x b128 loads in the GEMMs.
// transposed=1 means src is stored [N x K] (used for U: h_new = z_final U^T).
// ---------------------------------------------------------------------------
__global__ __launch_bounds__(256) void k_pack(const float* __restrict__ src,
                                              u16* __restrict__ dst,
                                              int K, int N, int batchStride,
                                              int nBatch, int transposed) {
  int idx = blockIdx.x * 256 + threadIdx.x;
  int kt_n = K >> 5, nt_n = N >> 4;
  int total = nBatch * nt_n * kt_n * 32;
  if (idx >= total) return;
  int lane = idx & 31;
  int t = idx >> 5;
  int kt = t % kt_n; t /= kt_n;
  int nt = t % nt_n; t /= nt_n;
  const float* s = src + (size_t)t * batchStride;
  int n  = nt * 16 + (lane & 15);
  int kb = kt * 32 + (lane >> 4) * 8;
  u16* d = dst + (size_t)idx * 16;
#pragma unroll
  for (int j = 0; j < 16; ++j) {
    int k = kb + (j < 8 ? j : j + 8);
    float v = transposed ? s[(size_t)n * K + k] : s[(size_t)k * N + n];
    d[j] = f2bf(v);
  }
}

// ---------------------------------------------------------------------------
// GEMM1: z[BTOK x 128] = h[BTOK x 2048] * V[2048 x 128]
// block = 64 tokens x 128 cols, 8 waves: (mw 0..3) x (nw 0..1), wave = 16x64.
// h staged f32->bf16 through LDS each 32-wide K step.
// ---------------------------------------------------------------------------
__global__ __launch_bounds__(256) void k_gemm_z(const float* __restrict__ h,
                                                const u16* __restrict__ Vp,
                                                float* __restrict__ zout) {
  __shared__ __align__(16) u16 shA[64 * 40];   // 64 rows x 32 halves, stride 40
  int tid = threadIdx.x;
  int wave = tid >> 5, lane = tid & 31;
  int mw = wave >> 1, nw = wave & 1;
  int m0 = blockIdx.x * 64;

  v8f acc[4];
#pragma unroll
  for (int i = 0; i < 4; ++i) acc[i] = v8f_zero();

  int arow = mw * 16 + (lane & 15);
  int hi8  = (lane >> 4) * 8;                  // A-frag K interleave offset

  int sr = tid >> 3;                           // staging: row 0..31 (+32)
  int sk = (tid & 7) * 4;                      // 4-float column chunk

  for (int kt = 0; kt < 64; ++kt) {
    int k0 = kt * 32;
    // cooperative stage h[64 x 32] f32 -> bf16 LDS
    {
      F4 f0 = *(const F4*)(h + (size_t)(m0 + sr) * DIM + k0 + sk);
      F4 f1 = *(const F4*)(h + (size_t)(m0 + sr + 32) * DIM + k0 + sk);
      u16* p0 = &shA[sr * 40 + sk];
      p0[0] = f2bf(f0.x); p0[1] = f2bf(f0.y); p0[2] = f2bf(f0.z); p0[3] = f2bf(f0.w);
      u16* p1 = &shA[(sr + 32) * 40 + sk];
      p1[0] = f2bf(f1.x); p1[1] = f2bf(f1.y); p1[2] = f2bf(f1.z); p1[3] = f2bf(f1.w);
      if (kt + 1 < 64)  // CDNA5 global_prefetch of next K tile
        __builtin_prefetch(h + (size_t)(m0 + sr) * DIM + k0 + 32 + sk, 0, 0);
    }
    __syncthreads();
    Frag A;
    const u16* ap = &shA[arow * 40 + hi8];
    A.q[0] = *(const U4*)ap;
    A.q[1] = *(const U4*)(ap + 16);
#pragma unroll
    for (int i = 0; i < 4; ++i) {
      int nt = nw * 4 + i;
      const U4* bp = (const U4*)Vp + (size_t)((nt * 64 + kt) * 32 + lane) * 2;
      Frag Bf; Bf.q[0] = bp[0]; Bf.q[1] = bp[1];
      acc[i] = wmma_bf16(A, Bf, acc[i]);
    }
    __syncthreads();
  }
  // C layout: lane -> col = lane&15, rows = j + (lane>=16 ? 8 : 0)
  int ncol = lane & 15, mrow = m0 + mw * 16 + hi8;
#pragma unroll
  for (int i = 0; i < 4; ++i) {
    int n0 = (nw * 4 + i) * 16;
#pragma unroll
    for (int j = 0; j < 8; ++j)
      zout[(size_t)(mrow + j) * RR + n0 + ncol] = acc[i][j];
  }
}

// ---------------------------------------------------------------------------
// Router: logits = z @ Ws, softmax, top-2 -> probs, combine weights w, blend.
// One thread per token.
// ---------------------------------------------------------------------------
__global__ __launch_bounds__(256) void k_router(const float* __restrict__ zbuf,
                                                const float* __restrict__ Ws,
                                                float* __restrict__ probs,
                                                float* __restrict__ wbuf,
                                                float* __restrict__ blend) {
  __shared__ float sWs[RR * EE];
  for (int i = threadIdx.x; i < RR * EE; i += 256) sWs[i] = Ws[i];
  __syncthreads();
  int t = blockIdx.x * 256 + threadIdx.x;
  const float* zp = zbuf + (size_t)t * RR;
  float lg[EE];
#pragma unroll
  for (int e = 0; e < EE; ++e) lg[e] = 0.f;
  for (int r = 0; r < RR; ++r) {
    float zv = zp[r];
#pragma unroll
    for (int e = 0; e < EE; ++e) lg[e] += zv * sWs[r * EE + e];
  }
  float mx = lg[0];
#pragma unroll
  for (int e = 1; e < EE; ++e) mx = fmaxf(mx, lg[e]);
  float p[EE], sum = 0.f;
#pragma unroll
  for (int e = 0; e < EE; ++e) { p[e] = expf(lg[e] - mx); sum += p[e]; }
  float inv = 1.f / sum;
#pragma unroll
  for (int e = 0; e < EE; ++e) { p[e] *= inv; probs[(size_t)t * EE + e] = p[e]; }
  int i1 = 0; float v1 = p[0];
#pragma unroll
  for (int e = 1; e < EE; ++e) if (p[e] > v1) { v1 = p[e]; i1 = e; }
  int i2 = (i1 == 0) ? 1 : 0; float v2 = p[i2];
#pragma unroll
  for (int e = 0; e < EE; ++e) if (e != i1 && p[e] > v2) { v2 = p[e]; i2 = e; }
  float w[EE];
#pragma unroll
  for (int e = 0; e < EE; ++e) w[e] = 0.f;
  w[i1] = v1; w[i2] = v2;
#pragma unroll
  for (int e = 0; e < EE; ++e) wbuf[(size_t)t * EE + e] = w[e];
  blend[t] = fminf(v1 + v2, 0.9f);
}

// ---------------------------------------------------------------------------
// Experts: per 16-token block, wave e computes
//   hid = silu(z W1[e] + b1[e]);  out_e = hid W2[e] + b2[e]
// H processed in 64-wide chunks: first-GEMM C-fragments -> silu -> bf16 LDS
// re-layout -> A-fragments of second GEMM. Ordered (deterministic) combine,
// then blend with z; z_final written in place over the z slice of d_out.
// ---------------------------------------------------------------------------
__global__ __launch_bounds__(256) void k_experts(
    const u16* __restrict__ W1p, const float* __restrict__ b1,
    const u16* __restrict__ W2p, const float* __restrict__ b2,
    const float* __restrict__ wbuf, const float* __restrict__ blend,
    float* __restrict__ zio /* in: z, out: z_final */) {
  __shared__ __align__(16) u16 zbf[16 * 136];
  __shared__ float zf32[16 * 128];
  __shared__ __align__(16) u16 hidb[8 * 16 * 72];
  __shared__ float znew[16 * 128];

  int tid = threadIdx.x;
  int wave = tid >> 5, lane = tid & 31;
  int t0 = blockIdx.x * 16;
  int e = wave;

  for (int i = tid; i < 16 * 128; i += 256) {
    int r = i >> 7, c = i & 127;
    float v = zio[(size_t)(t0 + r) * RR + c];
    zf32[i] = v;
    zbf[r * 136 + c] = f2bf(v);
    znew[i] = 0.f;
  }
  __syncthreads();

  int ncol = lane & 15;
  int hi8  = (lane >> 4) * 8;   // A-frag K offset == C-frag row offset

  Frag Az[4];                   // z A-fragments, reused across all H chunks
#pragma unroll
  for (int kt = 0; kt < 4; ++kt) {
    const u16* ap = &zbf[(lane & 15) * 136 + kt * 32 + hi8];
    Az[kt].q[0] = *(const U4*)ap;
    Az[kt].q[1] = *(const U4*)(ap + 16);
  }

  v8f outacc[8];
#pragma unroll
  for (int n = 0; n < 8; ++n) outacc[n] = v8f_zero();

  u16* myhid = &hidb[e * 16 * 72];

  for (int nc = 0; nc < 8; ++nc) {          // H chunk of 64
    v8f hac[4];
#pragma unroll
    for (int i = 0; i < 4; ++i) hac[i] = v8f_zero();
#pragma unroll
    for (int kt = 0; kt < 4; ++kt) {
#pragma unroll
      for (int i = 0; i < 4; ++i) {
        int nt = nc * 4 + i;
        const U4* bp = (const U4*)W1p +
                       (size_t)((e * 128 + nt * 4 + kt) * 32 + lane) * 2;
        Frag Bf; Bf.q[0] = bp[0]; Bf.q[1] = bp[1];
        hac[i] = wmma_bf16(Az[kt], Bf, hac[i]);
      }
    }
    // bias + silu, restage as bf16 A-operand rows (per-wave LDS, no barrier)
#pragma unroll
    for (int i = 0; i < 4; ++i) {
      float bb = b1[e * HH + nc * 64 + i * 16 + ncol];
#pragma unroll
      for (int j = 0; j < 8; ++j) {
        float x = hac[i][j] + bb;
        float s = x / (1.f + expf(-x));     // silu
        myhid[(j + hi8) * 72 + i * 16 + ncol] = f2bf(s);
      }
    }
#pragma unroll
    for (int kt2 = 0; kt2 < 2; ++kt2) {
      Frag A2;
      const u16* ap = &myhid[(lane & 15) * 72 + kt2 * 32 + hi8];
      A2.q[0] = *(const U4*)ap;
      A2.q[1] = *(const U4*)(ap + 16);
      int kt = nc * 2 + kt2;
#pragma unroll
      for (int n = 0; n < 8; ++n) {
        const U4* bp = (const U4*)W2p +
                       (size_t)((e * 128 + n * 16 + kt) * 32 + lane) * 2;
        Frag Bf; Bf.q[0] = bp[0]; Bf.q[1] = bp[1];
        outacc[n] = wmma_bf16(A2, Bf, outacc[n]);
      }
    }
  }

  float wrow[8];
#pragma unroll
  for (int j = 0; j < 8; ++j)
    wrow[j] = wbuf[(size_t)(t0 + j + hi8) * EE + e];

  // deterministic cross-expert combine (barrier-ordered, no float atomics)
  for (int p = 0; p < 8; ++p) {
    if (wave == p) {
#pragma unroll
      for (int n = 0; n < 8; ++n) {
        float bb = b2[e * RR + n * 16 + ncol];
#pragma unroll
        for (int j = 0; j < 8; ++j)
          znew[(j + hi8) * 128 + n * 16 + ncol] += (outacc[n][j] + bb) * wrow[j];
      }
    }
    __syncthreads();
  }

  for (int i = tid; i < 16 * 128; i += 256) {
    int r = i >> 7, c = i & 127;
    float bl = blend[t0 + r];
    zio[(size_t)(t0 + r) * RR + c] = znew[i] * bl + zf32[i] * (1.f - bl);
  }
}

// ---------------------------------------------------------------------------
// Reconstruct: h_new = z_final U^T (K=128, N=2048) fused with halt head.
// 16 tokens/block, wave covers a 256-wide N chunk (16 subtiles).
// ---------------------------------------------------------------------------
__global__ __launch_bounds__(256) void k_recon(const float* __restrict__ zfin,
                                               const u16* __restrict__ Up,
                                               const float* __restrict__ Wh,
                                               const float* __restrict__ bh,
                                               float* __restrict__ hnew,
                                               float* __restrict__ phalt) {
  __shared__ __align__(16) u16 zbf[16 * 136];
  __shared__ float haltp[8 * 16];

  int tid = threadIdx.x;
  int wave = tid >> 5, lane = tid & 31;
  int t0 = blockIdx.x * 16;

  for (int i = tid; i < 16 * 128; i += 256) {
    int r = i >> 7, c = i & 127;
    zbf[r * 136 + c] = f2bf(zfin[(size_t)(t0 + r) * RR + c]);
  }
  __syncthreads();

  int ncol = lane & 15;
  int hi8  = (lane >> 4) * 8;

  Frag Az[4];
#pragma unroll
  for (int kt = 0; kt < 4; ++kt) {
    const u16* ap = &zbf[(lane & 15) * 136 + kt * 32 + hi8];
    Az[kt].q[0] = *(const U4*)ap;
    Az[kt].q[1] = *(const U4*)(ap + 16);
  }

  float pr[8];
#pragma unroll
  for (int j = 0; j < 8; ++j) pr[j] = 0.f;

  for (int s = 0; s < 16; ++s) {
    int ntg = wave * 16 + s;                  // global 16-col tile
    v8f acc = v8f_zero();
#pragma unroll
    for (int kt = 0; kt < 4; ++kt) {
      const U4* bp = (const U4*)Up + (size_t)((ntg * 4 + kt) * 32 + lane) * 2;
      Frag Bf; Bf.q[0] = bp[0]; Bf.q[1] = bp[1];
      acc = wmma_bf16(Az[kt], Bf, acc);
    }
    int n0 = ntg * 16;
    float whv = Wh[n0 + ncol];
#pragma unroll
    for (int j = 0; j < 8; ++j) {
      float v = acc[j];
      hnew[(size_t)(t0 + j + hi8) * DIM + n0 + ncol] = v;
      pr[j] += v * whv;
    }
  }
  // reduce across the 16 lanes of each half-wave (columns of this chunk)
#pragma unroll
  for (int j = 0; j < 8; ++j) {
    float p = pr[j];
    p += __shfl_xor(p, 1);
    p += __shfl_xor(p, 2);
    p += __shfl_xor(p, 4);
    p += __shfl_xor(p, 8);
    pr[j] = p;
  }
  if (ncol == 0) {                            // lanes 0 and 16
#pragma unroll
    for (int j = 0; j < 8; ++j) haltp[wave * 16 + hi8 + j] = pr[j];
  }
  __syncthreads();
  if (tid < 16) {
    float s = 0.f;
#pragma unroll
    for (int w2 = 0; w2 < 8; ++w2) s += haltp[w2 * 16 + tid];
    phalt[t0 + tid] = 1.f / (1.f + expf(-(s + bh[0])));
  }
}

// ---------------------------------------------------------------------------
extern "C" void kernel_launch(void* const* d_in, const int* in_sizes, int n_in,
                              void* d_out, int out_size, void* d_ws,
                              size_t ws_size, hipStream_t stream) {
  (void)in_sizes; (void)n_in; (void)out_size; (void)ws_size;
  const float* h  = (const float*)d_in[0];
  const float* V  = (const float*)d_in[1];
  const float* U  = (const float*)d_in[2];
  const float* Ws = (const float*)d_in[3];
  const float* W1 = (const float*)d_in[4];
  const float* b1 = (const float*)d_in[5];
  const float* W2 = (const float*)d_in[6];
  const float* b2 = (const float*)d_in[7];
  const float* Wh = (const float*)d_in[8];
  const float* bh = (const float*)d_in[9];

  float* out = (float*)d_out;
  char*  ws  = (char*)d_ws;
  u16* Vp  = (u16*)(ws + WS_VPACK);
  u16* Up  = (u16*)(ws + WS_UPACK);
  u16* W1p = (u16*)(ws + WS_W1PACK);
  u16* W2p = (u16*)(ws + WS_W2PACK);
  float* wbuf  = (float*)(ws + WS_WBUF);
  float* blend = (float*)(ws + WS_BLEND);
  float* zio   = out + OUT_ZF;   // z lives here, overwritten by z_final

  // weight packing (bf16 B-fragment layout)
  k_pack<<<64,  256, 0, stream>>>(V,  Vp,  DIM, RR,  0,         1,  0);
  k_pack<<<64,  256, 0, stream>>>(U,  Up,  RR,  DIM, 0,         1,  1); // U is [N=D][K=R]
  k_pack<<<128, 256, 0, stream>>>(W1, W1p, RR,  HH,  RR * HH,   EE, 0);
  k_pack<<<128, 256, 0, stream>>>(W2, W2p, HH,  RR,  HH * RR,   EE, 0);

  k_gemm_z <<<BTOK / 64, 256, 0, stream>>>(h, Vp, zio);
  k_router <<<BTOK / 256, 256, 0, stream>>>(zio, Ws, out + OUT_PROBS, wbuf, blend);
  k_experts<<<BTOK / 16, 256, 0, stream>>>(W1p, b1, W2p, b2, wbuf, blend, zio);
  k_recon  <<<BTOK / 16, 256, 0, stream>>>(zio, Up, Wh, bh,
                                           out + OUT_HNEW, out + OUT_PHALT);
}
